// TrajEmbeddingNet_2920577761802
// MI455X (gfx1250) — matrix operand order repaired and checked by reading, not compiled, not run on hardware
//
#include <hip/hip_runtime.h>
#include <hip/hip_bf16.h>

typedef __attribute__((ext_vector_type(16))) __bf16 v16bf;
typedef __attribute__((ext_vector_type(8)))  __bf16 v8bf;
typedef __attribute__((ext_vector_type(8)))  float  v8f;

union V16U { v16bf v; v8bf h[2]; };

#define D 128
#define H 512
#define L 128
#define SEGS 512

// LDS byte offsets
#define LDS_W1T   0            // [512][128] bf16 = 131072 B  (W1^T: [hcol][k])
#define LDS_W2T   131072       // [128][512] bf16 = 131072 B  (W2^T: [ecol][k])
#define LDS_B1    262144       // 512 f32 = 2048 B
#define LDS_B2    264192       // 128 f32 = 512 B
#define LDS_TOTAL 264704

// ---------------------------------------------------------------------------
// prep: W1 [128][512] f32 -> W1T [512][128] bf16 ; W2 [512][128] f32 ->
// W2T [128][512] bf16 ; zero the segment-max buffer. All three are 65536 elems.
// ---------------------------------------------------------------------------
__global__ void prep_kernel(const float* __restrict__ W1, const float* __restrict__ W2,
                            __bf16* __restrict__ w1t, __bf16* __restrict__ w2t,
                            float* __restrict__ seg) {
  const int tid = blockIdx.x * blockDim.x + threadIdx.x;   // 0..65535
  { const int n = tid >> 7, k = tid & 127; w1t[tid] = (__bf16)W1[k * H + n]; }
  { const int n = tid >> 9, k = tid & 511; w2t[tid] = (__bf16)W2[k * L + n]; }
  seg[tid] = 0.0f;
}

// ---------------------------------------------------------------------------
// Fused MLP (layer1 + ReLU + layer2 + ReLU) + segment max via i32 atomics.
// One wave per 32-row tile (two 16-row WMMA tiles), 8 waves = 256 rows per WG.
//
// Layer 1 is computed TRANSPOSED:  Z = W1^T_tile (A, from LDS) x feat^T (B).
// Z's WMMA D-layout (lane = feat row, VGPR = h column) is bit-for-bit the
// A-operand layout layer 2 wants, so bias+ReLU+bf16-cvt happen purely in
// registers. Every LDS-sourced fragment (layer-1 A, layer-2 B) feeds TWO
// WMMAs (one per register-resident feat tile), halving LDS bytes per WMMA.
// ---------------------------------------------------------------------------
__global__ void __launch_bounds__(256, 1)
traj_fused(const float* __restrict__ feat, const int* __restrict__ tindex,
           const void* __restrict__ wts,          // w1t|w2t bf16, 256 KB
           const float* __restrict__ b1g, const float* __restrict__ b2g,
           float* __restrict__ seg, int Nrows) {
  extern __shared__ char smem[];
  __bf16* w1t = (__bf16*)(smem + LDS_W1T);
  __bf16* w2t = (__bf16*)(smem + LDS_W2T);
  float*  b1s = (float*)(smem + LDS_B1);
  float*  b2s = (float*)(smem + LDS_B2);

  // Stage 256 KB of bf16 weights + biases into LDS once per workgroup.
  {
    const uint4* src = (const uint4*)wts;
    uint4* dst = (uint4*)smem;
    for (int i = threadIdx.x; i < 16384; i += 256) dst[i] = src[i];
    if (threadIdx.x < 512) b1s[threadIdx.x] = b1g[threadIdx.x];
    if (threadIdx.x < 128) b2s[threadIdx.x] = b2g[threadIdx.x];
  }
  __syncthreads();

  const int lane = threadIdx.x & 31;
  const int lm = lane & 15;     // tile row/col selector
  const int lh = lane >> 4;     // lane-half selector
  const int rowBase = blockIdx.x * 256 + (threadIdx.x >> 5) * 32;
  if (rowBase + 32 > Nrows) return;   // wave-uniform

  // Preload feat^T B-fragments for BOTH 16-row tiles (f32 -> bf16), reused
  // across all 16 kk chunks. B 32x16 layout: lane = column N = feat row (lm);
  // elems 0..15 = contiguous K, chunk offset lh*16: one 64 B read per lane.
  V16U fb0[4], fb1[4];
  const float* frow0 = feat + (size_t)(rowBase + lm) * D;
  const float* frow1 = feat + (size_t)(rowBase + 16 + lm) * D;
  #pragma unroll
  for (int ks = 0; ks < 4; ++ks) {
    const int kb = ks * 32 + lh * 16;
    #pragma unroll
    for (int i = 0; i < 8; ++i) {
      fb0[ks].h[0][i] = (__bf16)frow0[kb + i];
      fb0[ks].h[1][i] = (__bf16)frow0[kb + 8 + i];
      fb1[ks].h[0][i] = (__bf16)frow1[kb + i];
      fb1[ks].h[1][i] = (__bf16)frow1[kb + 8 + i];
    }
  }

  v8f eacc0[8] = {};   // layer2 accumulators, feat rows [rowBase, +16)
  v8f eacc1[8] = {};   // layer2 accumulators, feat rows [rowBase+16, +32)

  for (int kk = 0; kk < 16; ++kk) {
    // ---- layer 1 (transposed): Z tiles for h-cols [kk*32, kk*32+32) ----
    v8f z00 = {}, z01 = {}, z10 = {}, z11 = {};
    #pragma unroll
    for (int ks = 0; ks < 4; ++ks) {
      // A = W1^T tile: lane holds h-col row (tile*16+lm), two 16 B K-chunks.
      V16U wa0, wa1;
      const __bf16* r0 = w1t + ((2 * kk)     * 16 + lm) * D + ks * 32 + lh * 16;
      const __bf16* r1 = w1t + ((2 * kk + 1) * 16 + lm) * D + ks * 32 + lh * 16;
      wa0.h[0] = *(const v8bf*)(r0);
      wa0.h[1] = *(const v8bf*)(r0 + 8);
      wa1.h[0] = *(const v8bf*)(r1);
      wa1.h[1] = *(const v8bf*)(r1 + 8);
      // Each LDS weight fragment feeds both feat tiles.
      z00 = __builtin_amdgcn_wmma_f32_16x16x32_bf16(false, wa0.v, false, fb0[ks].v,
                                                    (short)0, z00, false, false);
      z01 = __builtin_amdgcn_wmma_f32_16x16x32_bf16(false, wa1.v, false, fb0[ks].v,
                                                    (short)0, z01, false, false);
      z10 = __builtin_amdgcn_wmma_f32_16x16x32_bf16(false, wa0.v, false, fb1[ks].v,
                                                    (short)0, z10, false, false);
      z11 = __builtin_amdgcn_wmma_f32_16x16x32_bf16(false, wa1.v, false, fb1[ks].v,
                                                    (short)0, z11, false, false);
    }
    // ---- bias + ReLU + cvt: registers only; Z layout == layer-2 A layout ----
    const v8f bias0 = *(const v8f*)(b1s + kk * 32 + lh * 8);
    const v8f bias1 = *(const v8f*)(b1s + kk * 32 + 16 + lh * 8);
    V16U a2_0, a2_1;
    #pragma unroll
    for (int r = 0; r < 8; ++r) {
      a2_0.h[0][r] = (__bf16)fmaxf(z00[r] + bias0[r], 0.0f);
      a2_0.h[1][r] = (__bf16)fmaxf(z01[r] + bias1[r], 0.0f);
      a2_1.h[0][r] = (__bf16)fmaxf(z10[r] + bias0[r], 0.0f);
      a2_1.h[1][r] = (__bf16)fmaxf(z11[r] + bias1[r], 0.0f);
    }
    // ---- layer 2: accumulate this K=32 slab into all 8 output tiles ----
    #pragma unroll
    for (int nt = 0; nt < 8; ++nt) {
      V16U b2u;
      const __bf16* w2r = w2t + (nt * 16 + lm) * H + kk * 32 + lh * 8;
      b2u.h[0] = *(const v8bf*)(w2r);
      b2u.h[1] = *(const v8bf*)(w2r + 16);
      const v16bf b2f = b2u.v;
      // Each LDS W2 fragment feeds both feat tiles.
      eacc0[nt] = __builtin_amdgcn_wmma_f32_16x16x32_bf16(false, a2_0.v, false, b2f,
                                                          (short)0, eacc0[nt], false, false);
      eacc1[nt] = __builtin_amdgcn_wmma_f32_16x16x32_bf16(false, a2_1.v, false, b2f,
                                                          (short)0, eacc1[nt], false, false);
    }
  }

  // ---- epilogue: bias + ReLU, then segment max (i32 atomic; values >= 0) ----
  int sidx0[8], sidx1[8];
  #pragma unroll
  for (int r = 0; r < 8; ++r) {
    sidx0[r] = tindex[rowBase + lh * 8 + r];
    sidx1[r] = tindex[rowBase + 16 + lh * 8 + r];
  }
  #pragma unroll
  for (int nt = 0; nt < 8; ++nt) {
    const int col = nt * 16 + lm;
    const float bias = b2s[col];
    #pragma unroll
    for (int r = 0; r < 8; ++r) {
      const float v0 = fmaxf(eacc0[nt][r] + bias, 0.0f);
      const float v1 = fmaxf(eacc1[nt][r] + bias, 0.0f);
      atomicMax((int*)seg + sidx0[r] * L + col, __float_as_int(v0));
      atomicMax((int*)seg + sidx1[r] * L + col, __float_as_int(v1));
    }
  }
}

// ---------------------------------------------------------------------------
// Layer 3 (tiny, exact f32): out = seg @ W3 + b3, [512,128]@[128,128]
// ---------------------------------------------------------------------------
__global__ void layer3_kernel(const float* __restrict__ seg, const float* __restrict__ W3,
                              const float* __restrict__ b3, float* __restrict__ out) {
  const int tid = blockIdx.x * blockDim.x + threadIdx.x;   // 0..65535
  const int s = tid >> 7, j = tid & 127;
  float acc = b3[j];
  const float* sr = seg + s * L;
  #pragma unroll 4
  for (int k = 0; k < 128; ++k) acc = fmaf(sr[k], W3[k * L + j], acc);
  out[tid] = acc;
}

extern "C" void kernel_launch(void* const* d_in, const int* in_sizes, int n_in,
                              void* d_out, int out_size, void* d_ws, size_t ws_size,
                              hipStream_t stream) {
  const float* feat = (const float*)d_in[0];
  const int*   tidx = (const int*)d_in[1];
  const float* W1   = (const float*)d_in[2];
  const float* b1   = (const float*)d_in[3];
  const float* W2   = (const float*)d_in[4];
  const float* b2   = (const float*)d_in[5];
  const float* W3   = (const float*)d_in[6];
  const float* b3   = (const float*)d_in[7];
  float* out = (float*)d_out;

  const int N = in_sizes[0] / D;
  char* ws = (char*)d_ws;
  __bf16* w1t = (__bf16*)(ws);                // 131072 B
  __bf16* w2t = (__bf16*)(ws + 131072);       // 131072 B
  float*  seg = (float*)(ws + 262144);        // 262144 B

  // Allow >64KB dynamic LDS on CDNA5 (320KB/WGP). Deterministic, host-side.
  static_cast<void>(hipFuncSetAttribute(
      reinterpret_cast<const void*>(traj_fused),
      hipFuncAttributeMaxDynamicSharedMemorySize, LDS_TOTAL));

  prep_kernel<<<256, 256, 0, stream>>>(W1, W2, w1t, w2t, seg);
  const int grid = (N + 255) / 256;
  traj_fused<<<grid, 256, LDS_TOTAL, stream>>>(feat, tidx, (const void*)w1t, b1, b2, seg, N);
  layer3_kernel<<<256, 256, 0, stream>>>(seg, W3, b3, out);
}
